// GATLayer_56418690400271
// MI455X (gfx1250) — compile-verified, matrix operand order checked by reading, not run on hardware
//
#include <hip/hip_runtime.h>

#define DIMC 128
#define HH   8
#define CC   16
#define NEG_SLOPE 0.2f
#define LN_EPS 1e-5f

typedef __attribute__((ext_vector_type(16))) _Float16 v16h;
typedef __attribute__((ext_vector_type(8)))  float    v8f;

// ---------- helpers ----------
__device__ __forceinline__ unsigned enc_f32(float f) {
    unsigned u = __float_as_uint(f);
    return (u >> 31) ? ~u : (u | 0x80000000u);   // monotone order-preserving map
}
__device__ __forceinline__ float dec_f32(unsigned u) {
    return __uint_as_float((u & 0x80000000u) ? (u & 0x7FFFFFFFu) : ~u);
}

// Preload ALL four 16x32 f16 A-fragments (ISA 7.12.2 layout) for a 128-float
// row: 16 float4 global loads issued up-front, then converted to f16.
__device__ __forceinline__ void preload_a_frags(const float* __restrict__ Arow,
                                                int half, v16h afrag[4]) {
    float4 p[16];
#pragma unroll
    for (int kt = 0; kt < 4; ++kt) {
        int base = kt * 32 + half * 8;
        p[4 * kt + 0] = *(const float4*)(Arow + base);
        p[4 * kt + 1] = *(const float4*)(Arow + base + 4);
        p[4 * kt + 2] = *(const float4*)(Arow + base + 16);
        p[4 * kt + 3] = *(const float4*)(Arow + base + 20);
    }
#pragma unroll
    for (int kt = 0; kt < 4; ++kt) {
        v16h a;
#pragma unroll
        for (int q = 0; q < 4; ++q) {
            float4 f = p[4 * kt + q];
            a[4 * q + 0] = (_Float16)f.x;
            a[4 * q + 1] = (_Float16)f.y;
            a[4 * q + 2] = (_Float16)f.z;
            a[4 * q + 3] = (_Float16)f.w;
        }
        afrag[kt] = a;
    }
}

// Load a 32x16 f16 B-fragment from LDS holding W transposed (Wt[n*128+k]).
__device__ __forceinline__ v16h load_b_frag_lds(const _Float16* __restrict__ Wn,
                                                int kt, int half) {
    v16h b;
#pragma unroll
    for (int j = 0; j < 8; ++j) {
        int K0 = kt * 32 + (j >> 2) * 16 + half * 8 + (j & 3) * 2;
        b[2 * j]     = Wn[K0];       // contiguous pair -> paired ds_load_b128
        b[2 * j + 1] = Wn[K0 + 1];
    }
    return b;
}

// ---------- 1) LayerNorm: one wave32 per row ----------
__global__ __launch_bounds__(256) void k_layernorm(const float* __restrict__ x,
                                                   const float* __restrict__ gamma,
                                                   const float* __restrict__ beta,
                                                   float* __restrict__ xn, int N) {
    int wave = threadIdx.x >> 5, lane = threadIdx.x & 31;
    int row = blockIdx.x * 8 + wave;
    if (row >= N) return;
    float4 v = ((const float4*)(x + (size_t)row * DIMC))[lane];
    float s  = v.x + v.y + v.z + v.w;
    float sq = v.x * v.x + v.y * v.y + v.z * v.z + v.w * v.w;
#pragma unroll
    for (int m = 1; m < 32; m <<= 1) {
        s  += __shfl_xor(s, m);
        sq += __shfl_xor(sq, m);
    }
    float mean = s * (1.0f / DIMC);
    float var  = sq * (1.0f / DIMC) - mean * mean;
    float rs   = rsqrtf(var + LN_EPS);
    float4 g  = ((const float4*)gamma)[lane];
    float4 bt = ((const float4*)beta)[lane];
    float4 o;
    o.x = (v.x - mean) * rs * g.x + bt.x;
    o.y = (v.y - mean) * rs * g.y + bt.y;
    o.z = (v.z - mean) * rs * g.z + bt.z;
    o.w = (v.w - mean) * rs * g.w + bt.w;
    ((float4*)(xn + (size_t)row * DIMC))[lane] = o;
}

// ---------- 2) dual GEMM: Y1 = A@W1+b1, Y2 = A@W2+b2 (M x 128 x 128) --------
// Shares A fragments between both weight matrices; both W staged transposed
// as f16 in LDS (64 KB total). Block = 8 waves, 128 rows per block.
__global__ __launch_bounds__(256) void k_gemm_dual(const float* __restrict__ A,
                                                   const float* __restrict__ W1,
                                                   const float* __restrict__ b1,
                                                   const float* __restrict__ W2,
                                                   const float* __restrict__ b2,
                                                   float* __restrict__ Y1,
                                                   float* __restrict__ Y2, int M) {
    __shared__ _Float16 Wt1[DIMC * DIMC];   // Wt[n*128 + k]
    __shared__ _Float16 Wt2[DIMC * DIMC];
    int tid = threadIdx.x;
    for (int i = tid; i < DIMC * DIMC; i += 256) {
        int k = i >> 7, n = i & 127;
        Wt1[n * DIMC + k] = (_Float16)W1[i];
        Wt2[n * DIMC + k] = (_Float16)W2[i];
    }
    __syncthreads();

    int wave = tid >> 5, lane = tid & 31, half = lane >> 4;
    int rowBase = blockIdx.x * 128 + wave * 16;
    int r  = rowBase + (lane & 15);
    int rc = r < M ? r : (M - 1);              // clamp tail loads, mask stores
    const float* Arow = A + (size_t)rc * DIMC;

    v16h afrag[4];
    preload_a_frags(Arow, half, afrag);

    v8f acc1[8] = {};
    v8f acc2[8] = {};
#pragma unroll
    for (int kt = 0; kt < 4; ++kt) {
#pragma unroll
        for (int nt = 0; nt < 8; ++nt) {
            int n = nt * 16 + (lane & 15);
            v16h b1f = load_b_frag_lds(&Wt1[n * DIMC], kt, half);
            acc1[nt] = __builtin_amdgcn_wmma_f32_16x16x32_f16(
                false, afrag[kt], false, b1f, (short)0, acc1[nt], false, false);
            v16h b2f = load_b_frag_lds(&Wt2[n * DIMC], kt, half);
            acc2[nt] = __builtin_amdgcn_wmma_f32_16x16x32_f16(
                false, afrag[kt], false, b2f, (short)0, acc2[nt], false, false);
        }
    }
#pragma unroll
    for (int nt = 0; nt < 8; ++nt) {
        int col   = nt * 16 + (lane & 15);
        float bv1 = b1[col];
        float bv2 = b2[col];
#pragma unroll
        for (int j = 0; j < 8; ++j) {
            int row = rowBase + half * 8 + j;   // D layout: M = vgpr + 8*(lane/16)
            if (row < M) {
                Y1[(size_t)row * DIMC + col] = acc1[nt][j] + bv1;
                Y2[(size_t)row * DIMC + col] = acc2[nt][j] + bv2;
            }
        }
    }
}

// ---------- 3) init: out = bias broadcast, emax/denom cleared ----------
__global__ void k_init(float* __restrict__ out, const float* __restrict__ bias,
                       unsigned* __restrict__ emax, float* __restrict__ denom,
                       int total, int nh) {
    int idx = blockIdx.x * blockDim.x + threadIdx.x;
    if (idx < total) out[idx] = bias[idx & 127];
    if (idx < nh) { emax[idx] = 0u; denom[idx] = 0.0f; }
}

// ---------- 4) fused ee-GEMM + gather + leakyReLU + att-dot -> edge logits ----
// Wave handles 16 edges; each WMMA N-tile == one head (C==16). ee never stored.
__global__ __launch_bounds__(256) void k_edge_logits(
    const float* __restrict__ EA, const float* __restrict__ We,
    const float* __restrict__ be, const float* __restrict__ xl,
    const float* __restrict__ xr, const long long* __restrict__ src,
    const long long* __restrict__ dst, const float* __restrict__ att,
    float* __restrict__ elog, int E) {
    __shared__ _Float16 Wt[DIMC * DIMC];
    int tid = threadIdx.x;
    for (int i = tid; i < DIMC * DIMC; i += 256) {
        int k = i >> 7, n = i & 127;
        Wt[n * DIMC + k] = (_Float16)We[i];
    }
    __syncthreads();

    int wave = tid >> 5, lane = tid & 31, half = lane >> 4;
    int eBase = (blockIdx.x * 8 + wave) * 16;
    int er  = eBase + (lane & 15);
    int erc = er < E ? er : (E - 1);
    const float* Arow = EA + (size_t)erc * DIMC;

    v16h afrag[4];
    preload_a_frags(Arow, half, afrag);

    // D-tile rows for this lane: edges eBase + half*8 + j
    long long sj[8], dj[8];
#pragma unroll
    for (int j = 0; j < 8; ++j) {
        int row = eBase + half * 8 + j;
        int rcl = row < E ? row : (E - 1);
        sj[j] = src[rcl];
        dj[j] = dst[rcl];
    }

    v8f acc[8] = {};
#pragma unroll
    for (int kt = 0; kt < 4; ++kt) {
#pragma unroll
        for (int nt = 0; nt < 8; ++nt) {
            int n = nt * 16 + (lane & 15);
            v16h b = load_b_frag_lds(&Wt[n * DIMC], kt, half);
            acc[nt] = __builtin_amdgcn_wmma_f32_16x16x32_f16(
                false, afrag[kt], false, b, (short)0, acc[nt], false, false);
        }
    }

#pragma unroll
    for (int nt = 0; nt < 8; ++nt) {            // nt == head h (C==16)
        int c   = lane & 15;
        int col = nt * 16 + c;                  // channel index in DIM
        float av  = att[col];                   // att[h*16+c]
        float bev = be[col];
        float partial[8];
#pragma unroll
        for (int j = 0; j < 8; ++j) {
            float m = acc[nt][j] + bev
                    + xl[(size_t)sj[j] * DIMC + col]
                    + xr[(size_t)dj[j] * DIMC + col];
            float lr = m > 0.0f ? m : NEG_SLOPE * m;
            partial[j] = lr * av;
        }
#pragma unroll
        for (int j = 0; j < 8; ++j) {           // reduce over the 16 channels
            float v = partial[j];
            v += __shfl_xor(v, 1);
            v += __shfl_xor(v, 2);
            v += __shfl_xor(v, 4);
            v += __shfl_xor(v, 8);
            partial[j] = v;
        }
        if ((lane & 15) == 0) {
#pragma unroll
            for (int j = 0; j < 8; ++j) {
                int row = eBase + half * 8 + j;
                if (row < E) elog[(size_t)row * HH + nt] = partial[j];
            }
        }
    }
}

// ---------- 5) segment max via order-preserving uint atomicMax ----------
__global__ void k_segmax(const float* __restrict__ elog,
                         const long long* __restrict__ dst,
                         unsigned* __restrict__ emax, int EH) {
    int idx = blockIdx.x * blockDim.x + threadIdx.x;
    if (idx >= EH) return;
    int e = idx >> 3, h = idx & 7;
    long long d = dst[e];
    atomicMax(&emax[(size_t)d * HH + h], enc_f32(elog[idx]));
}

// ---------- 6) exp(e - max) in place + segment sum of denominators ----------
__global__ void k_expsum(float* __restrict__ elog,
                         const long long* __restrict__ dst,
                         const unsigned* __restrict__ emax,
                         float* __restrict__ denom, int EH) {
    int idx = blockIdx.x * blockDim.x + threadIdx.x;
    if (idx >= EH) return;
    int e = idx >> 3, h = idx & 7;
    long long d = dst[e];
    float m = dec_f32(emax[(size_t)d * HH + h]);
    float v = __expf(elog[idx] - m);
    elog[idx] = v;
    atomicAdd(&denom[(size_t)d * HH + h], v);
}

// ---------- 7) scatter-aggregate alpha * xl[src] into out (L2-resident) ------
__global__ void k_aggregate(const float* __restrict__ elog,
                            const float* __restrict__ denom,
                            const float* __restrict__ xl,
                            const long long* __restrict__ src,
                            const long long* __restrict__ dst,
                            float* __restrict__ out, int total) {
    int idx = blockIdx.x * blockDim.x + threadIdx.x;
    if (idx >= total) return;
    int e = idx >> 7, d = idx & 127, h = d >> 4;
    long long dn = dst[e], sn = src[e];
    float alpha = elog[(size_t)e * HH + h] / denom[(size_t)dn * HH + h];
    atomicAdd(&out[(size_t)dn * DIMC + d], alpha * xl[(size_t)sn * DIMC + d]);
}

// ---------- launch ----------
extern "C" void kernel_launch(void* const* d_in, const int* in_sizes, int n_in,
                              void* d_out, int out_size, void* d_ws, size_t ws_size,
                              hipStream_t stream) {
    const float*     x   = (const float*)d_in[0];
    const long long* ei  = (const long long*)d_in[1];   // int64 edge_index [2,E]
    const float*     ea  = (const float*)d_in[2];
    const float*     gam = (const float*)d_in[3];
    const float*     bet = (const float*)d_in[4];
    const float*     W_l = (const float*)d_in[5];
    const float*     b_l = (const float*)d_in[6];
    const float*     W_r = (const float*)d_in[7];
    const float*     b_r = (const float*)d_in[8];
    const float*     W_e = (const float*)d_in[9];
    const float*     b_e = (const float*)d_in[10];
    const float*     att = (const float*)d_in[11];
    const float*     bia = (const float*)d_in[12];
    float* out = (float*)d_out;

    const int N = in_sizes[0] / DIMC;
    const int E = in_sizes[1] / 2;
    const long long* src = ei;
    const long long* dst = ei + E;

    char* ws = (char*)d_ws;
    float* xn    = (float*)ws;  ws += (size_t)N * DIMC * sizeof(float);
    float* xl    = (float*)ws;  ws += (size_t)N * DIMC * sizeof(float);
    float* xr    = (float*)ws;  ws += (size_t)N * DIMC * sizeof(float);
    float* elog  = (float*)ws;  ws += (size_t)E * HH * sizeof(float);
    unsigned* emax = (unsigned*)ws; ws += (size_t)N * HH * sizeof(unsigned);
    float* denom = (float*)ws;  ws += (size_t)N * HH * sizeof(float);

    k_layernorm<<<(N + 7) / 8, 256, 0, stream>>>(x, gam, bet, xn, N);
    k_gemm_dual<<<(N + 127) / 128, 256, 0, stream>>>(xn, W_l, b_l, W_r, b_r,
                                                     xl, xr, N);
    k_init<<<((size_t)N * DIMC + 255) / 256, 256, 0, stream>>>(
        out, bia, emax, denom, N * DIMC, N * HH);
    k_edge_logits<<<(E + 127) / 128, 256, 0, stream>>>(
        ea, W_e, b_e, xl, xr, src, dst, att, elog, E);
    k_segmax<<<((size_t)E * HH + 255) / 256, 256, 0, stream>>>(elog, dst, emax, E * HH);
    k_expsum<<<((size_t)E * HH + 255) / 256, 256, 0, stream>>>(elog, dst, emax, denom, E * HH);
    k_aggregate<<<((size_t)E * DIMC + 255) / 256, 256, 0, stream>>>(
        elog, denom, xl, src, dst, out, E * DIMC);
}
